// SGC_38448547234512
// MI455X (gfx1250) — matrix operand order repaired and checked by reading, not compiled, not run on hardware
//
#include <hip/hip_runtime.h>
#include <hip/hip_bf16.h>

typedef __attribute__((ext_vector_type(16))) _Float16 v16h;
typedef __attribute__((ext_vector_type(8)))  float    v8f;

#define NNODES 100000
#define FEAT   64
#define HIDDEN 64
#define NCLS   40

// ---------------- degree / normalization ----------------

__global__ void sgc_deg_init(float* __restrict__ deg, int n) {
    int i = blockIdx.x * blockDim.x + threadIdx.x;
    if (i < n) deg[i] = 1.0f;   // self-loop contributes 1
}

__global__ void sgc_deg_accum(const int* __restrict__ dst, float* __restrict__ deg, int e) {
    int i = blockIdx.x * blockDim.x + threadIdx.x;
    if (i < e) atomicAdd(&deg[dst[i]], 1.0f);
}

__global__ void sgc_deg_rsqrt(float* __restrict__ deg, int n) {
    int i = blockIdx.x * blockDim.x + threadIdx.x;
    if (i < n) deg[i] = rsqrtf(deg[i]);  // deg >= 1 always (self-loop)
}

// ---------------- propagation hops ----------------

// y[i,f] = dinv[i]^2 * xin[i,f]   (self-loop term; also fully initializes y)
__global__ void sgc_prop_self(const float* __restrict__ xin,
                              const float* __restrict__ dinv,
                              float* __restrict__ y, int n) {
    int idx = blockIdx.x * blockDim.x + threadIdx.x;
    if (idx < n * FEAT) {
        int i = idx >> 6;
        float d = dinv[i];
        y[idx] = d * d * xin[idx];
    }
}

// y[dst,f] += dinv[src]*dinv[dst] * xin[src,f]  for every edge
__global__ void sgc_prop_scatter(const float* __restrict__ xin,
                                 const int* __restrict__ src,
                                 const int* __restrict__ dst,
                                 const float* __restrict__ dinv,
                                 float* __restrict__ y, int e) {
    int idx = blockIdx.x * blockDim.x + threadIdx.x;
    if (idx < e * FEAT) {
        int ee = idx >> 6;
        int f  = idx & 63;
        int s = src[ee], d = dst[ee];
        float w = dinv[s] * dinv[d];
        atomicAdd(&y[(size_t)d * FEAT + f], w * xin[(size_t)s * FEAT + f]);
    }
}

// ---------------- fused dense head: GEMM1+ReLU -> GEMM2 -> log_softmax ----------------
// One wave32 per 16-row tile. 100000 = 6250 * 16 exactly.
//
// WMMA 16x16x32 f16 fragment layouts used here (wave32, contiguous k order, same in A and B):
//   A (16x32, MxK): lane L holds row M = L&15; half j holds
//       K = j + (L>=16 ? 16 : 0)                                (+ kstep*32)
//   B (32x16, KxN): lane L holds col N = L&15; half j holds
//       K = j + (L>=16 ? 16 : 0)                                (+ kstep*32)
//   C/D (16x16 f32): lane L holds col N = L&15; float r holds row M = r + (L>=16 ? 8 : 0)

__global__ __launch_bounds__(32) void sgc_head(const float* __restrict__ xprop, // [N,64]
                                               const float* __restrict__ W1,    // [64,64]
                                               const float* __restrict__ b1,    // [64]
                                               const float* __restrict__ W2,    // [64,40]
                                               const float* __restrict__ b2,    // [40]
                                               float* __restrict__ out)         // [N,40]
{
    __shared__ float lds_h[16 * HIDDEN];  // h tile, row-major
    __shared__ float lds_lg[16 * 48];     // logits tile, padded to 48 cols

    const int lane   = threadIdx.x;      // 0..31
    const int mrow   = lane & 15;
    const int hiHalf = lane >> 4;        // 0 or 1
    const int row0   = blockIdx.x * 16;

    __builtin_prefetch(W1, 0, 0);
    __builtin_prefetch(W2, 0, 0);

    // ---- A fragments of the propagated-x tile (f32 -> f16), K = 0..31 and 32..63
    v16h a0, a1;
    const float* xr = xprop + (size_t)(row0 + mrow) * FEAT;
#pragma unroll
    for (int j = 0; j < 16; ++j) {
        int k = j + (hiHalf ? 16 : 0);
        a0[j] = (_Float16)xr[k];
        a1[j] = (_Float16)xr[32 + k];
    }

    // ---- GEMM1: h[16x64] = x[16x64] @ W1[64x64], 4 N-tiles x 2 K-steps
#pragma unroll
    for (int n = 0; n < 4; ++n) {
        v8f c = {};
        const int col = n * 16 + mrow;
#pragma unroll
        for (int ks = 0; ks < 2; ++ks) {
            v16h b;
            const int kb = ks * 32 + hiHalf * 16;
#pragma unroll
            for (int j = 0; j < 16; ++j)
                b[j] = (_Float16)W1[(kb + j) * HIDDEN + col];
            c = __builtin_amdgcn_wmma_f32_16x16x32_f16(
                    false, ks ? a1 : a0, false, b, (short)0, c, false, false);
        }
        // bias + ReLU, stash in LDS (C/D layout -> row-major)
        const float bias = b1[col];
#pragma unroll
        for (int r = 0; r < 8; ++r) {
            float v = c[r] + bias;
            v = v > 0.0f ? v : 0.0f;
            lds_h[(r + hiHalf * 8) * HIDDEN + col] = v;
        }
    }
    __syncthreads();

    // ---- re-fragment h as WMMA A operands
    v16h h0, h1;
#pragma unroll
    for (int j = 0; j < 16; ++j) {
        int k = j + (hiHalf ? 16 : 0);
        h0[j] = (_Float16)lds_h[mrow * HIDDEN + k];
        h1[j] = (_Float16)lds_h[mrow * HIDDEN + 32 + k];
    }

    // ---- GEMM2: logits[16x40] = h @ W2[64x40], 3 N-tiles (pad 40->48) x 2 K-steps
#pragma unroll
    for (int n = 0; n < 3; ++n) {
        v8f c = {};
        const int col = n * 16 + mrow;           // 0..47
#pragma unroll
        for (int ks = 0; ks < 2; ++ks) {
            v16h b;
            const int kb = ks * 32 + hiHalf * 16;
#pragma unroll
            for (int j = 0; j < 16; ++j) {
                float w = (col < NCLS) ? W2[(kb + j) * NCLS + col] : 0.0f;
                b[j] = (_Float16)w;
            }
            c = __builtin_amdgcn_wmma_f32_16x16x32_f16(
                    false, ks ? h1 : h0, false, b, (short)0, c, false, false);
        }
        const float bias = (col < NCLS) ? b2[col] : 0.0f;
#pragma unroll
        for (int r = 0; r < 8; ++r)
            lds_lg[(r + hiHalf * 8) * 48 + col] = c[r] + bias;
    }
    __syncthreads();

    // ---- log_softmax over 40 classes, one row per lane (lanes 0..15)
    if (lane < 16) {
        const float* lrow = lds_lg + lane * 48;
        float mx = -1e30f;
        for (int cc = 0; cc < NCLS; ++cc) mx = fmaxf(mx, lrow[cc]);
        float s = 0.0f;
        for (int cc = 0; cc < NCLS; ++cc) s += __expf(lrow[cc] - mx);
        const float lse = mx + __logf(s);
        float* orow = out + (size_t)(row0 + lane) * NCLS;
        for (int cc = 0; cc < NCLS; ++cc) orow[cc] = lrow[cc] - lse;
    }
}

// ---------------- launch ----------------

extern "C" void kernel_launch(void* const* d_in, const int* in_sizes, int n_in,
                              void* d_out, int out_size, void* d_ws, size_t ws_size,
                              hipStream_t stream) {
    const float* x  = (const float*)d_in[0];   // [N,64]
    const int*   ei = (const int*)d_in[1];     // [2,E]
    const float* W1 = (const float*)d_in[2];   // [64,64]
    const float* b1 = (const float*)d_in[3];   // [64]
    const float* W2 = (const float*)d_in[4];   // [64,40]
    const float* b2 = (const float*)d_in[5];   // [40]
    float* out = (float*)d_out;

    const int n = NNODES;
    const int e = in_sizes[1] / 2;
    const int* src = ei;
    const int* dst = ei + e;

    // workspace carve: dinv[N] | xbuf0[N*64] | xbuf1[N*64]
    char* ws = (char*)d_ws;
    float* dinv = (float*)ws;
    size_t off  = (((size_t)n * sizeof(float)) + 511) & ~(size_t)511;
    float* xb0  = (float*)(ws + off);
    float* xb1  = (float*)(ws + off + (size_t)n * FEAT * sizeof(float));

    const int T = 256;
    // normalization
    sgc_deg_init <<<(n + T - 1) / T, T, 0, stream>>>(dinv, n);
    sgc_deg_accum<<<(e + T - 1) / T, T, 0, stream>>>(dst, dinv, e);
    sgc_deg_rsqrt<<<(n + T - 1) / T, T, 0, stream>>>(dinv, n);

    const int nwork = n * FEAT;   // 6.4M
    const int ework = e * FEAT;   // 76.8M
    // hop 1: x -> xb0
    sgc_prop_self   <<<(nwork + T - 1) / T, T, 0, stream>>>(x, dinv, xb0, n);
    sgc_prop_scatter<<<(ework + T - 1) / T, T, 0, stream>>>(x, src, dst, dinv, xb0, e);
    // hop 2: xb0 -> xb1
    sgc_prop_self   <<<(nwork + T - 1) / T, T, 0, stream>>>(xb0, dinv, xb1, n);
    sgc_prop_scatter<<<(ework + T - 1) / T, T, 0, stream>>>(xb0, src, dst, dinv, xb1, e);

    // fused dense head: one wave per 16-row tile
    sgc_head<<<n / 16, 32, 0, stream>>>(xb1, W1, b1, W2, b2, out);
}
